// GraphHead_17781164605812
// MI455X (gfx1250) — compile-verified, matrix operand order checked by reading, not run on hardware
//
#include <hip/hip_runtime.h>
#include <math.h>

// ---------------------------------------------------------------------------
// Problem constants (from reference)
// ---------------------------------------------------------------------------
#define BATCH   64
#define SEQ     2048
#define FT_IN   768
#define DDIM    512
#define HCDIM   256
#define NHEAD   4
#define CDIM    64
#define NNODE   2049            // SEQ + 1
#define NROWS   (BATCH*SEQ)     // 131072
#define NTOT    (BATCH*NNODE)   // 131136

// ---------------------------------------------------------------------------
// Vector types
// ---------------------------------------------------------------------------
typedef __attribute__((ext_vector_type(16))) _Float16 v16h;
typedef __attribute__((ext_vector_type(8)))  _Float16 v8h;
typedef __attribute__((ext_vector_type(4)))  _Float16 v4h;
typedef __attribute__((ext_vector_type(8)))  float    v8f;
typedef __attribute__((ext_vector_type(4)))  float    v4f;

__device__ __forceinline__ float gelu_f(float x) {
    return 0.5f * x * (1.0f + erff(x * 0.70710678118654752f));
}
__device__ __forceinline__ float lrelu_f(float x) {
    return x > 0.0f ? x : 0.2f * x;
}
// LDS byte offset of a __shared__ pointer (generic ptr low 32 bits)
__device__ __forceinline__ unsigned lds_off(const void* p) {
    return (unsigned)(unsigned long long)(uintptr_t)p;
}

// ---------------------------------------------------------------------------
// Tiled GEMM:  C[M,N] = act(A[M,K] @ Wt[N,K]^T + bias[N])     (all f16 I/O)
// A: M x K f16 row-major.  Wt: N x K f16 (pre-transposed weight).
// Block = 256 threads (8 waves). Tile BM=128, BN=128, BK=32.
// Wave (wr=w&3, wc=w>>2) owns a 32x64 subtile = 2x4 WMMA 16x16 tiles.
// Staging: GLOBAL_LOAD_ASYNC_TO_LDS_B128 (16B/lane) into double-buffered
// LDS; next tile's async loads overlap current tile's WMMAs; sync via
// s_wait_asynccnt + workgroup barrier.  Row guard by index clamp.
// ---------------------------------------------------------------------------
#define BM 128
#define BN 128
#define BK 32
#define LDP 48   // LDS row stride in halfs: 96B (multiple of 16B)

template<int ACT>
__global__ __launch_bounds__(256) void gemm_wmma(
    const _Float16* __restrict__ A, const _Float16* __restrict__ Wt,
    const float* __restrict__ bias, _Float16* __restrict__ Cout,
    int M, int K, int N_)
{
    __shared__ __align__(16) _Float16 lA[2][BM][LDP];
    __shared__ __align__(16) _Float16 lB[2][BN][LDP];

    const int tid  = threadIdx.x;
    const int lane = tid & 31;
    const int w    = tid >> 5;
    const int wr   = w & 3;          // wave row  -> 32*wr
    const int wc   = w >> 2;         // wave col  -> 64*wc
    const int m0   = blockIdx.y * BM;
    const int n0   = blockIdx.x * BN;
    const int lrow  = lane & 15;
    const int khalf = (lane >> 4) * 8;   // 0 (lanes 0-15) or 8 (lanes 16-31)

    const unsigned bufStride = (unsigned)(BM * LDP * sizeof(_Float16)); // 12288B

    v8f acc[2][4];
    #pragma unroll
    for (int i = 0; i < 2; i++)
        #pragma unroll
        for (int j = 0; j < 4; j++)
            acc[i][j] = (v8f)(0.0f);

    // Each tile (128 rows x 32 halfs) = 512 x 16B transfers = 2 per thread.
    // idx = tid + i*256 ; r = idx>>2 ; c8 = idx&3 (8-half chunk within row)
    auto issueA = [&](int k0, int buf) {
        #pragma unroll
        for (int i = 0; i < 2; i++) {
            int idx = tid + i * 256;
            int r = idx >> 2, c8 = idx & 3;
            int gr  = m0 + r;
            int grc = gr < M ? gr : M - 1;              // clamp: always valid
            const _Float16* gp = A + (long)grc * K + k0 + c8 * 8;
            unsigned l = lds_off(&lA[0][r][c8 * 8]) + (unsigned)buf * bufStride;
            asm volatile("global_load_async_to_lds_b128 %0, %1, off"
                         :: "v"(l), "v"((unsigned long long)(uintptr_t)gp)
                         : "memory");
        }
    };
    auto issueB = [&](int k0, int buf) {
        #pragma unroll
        for (int i = 0; i < 2; i++) {
            int idx = tid + i * 256;
            int r = idx >> 2, c8 = idx & 3;             // r = N column of tile
            const _Float16* gp = Wt + (long)(n0 + r) * K + k0 + c8 * 8;
            unsigned l = lds_off(&lB[0][r][c8 * 8]) + (unsigned)buf * bufStride;
            asm volatile("global_load_async_to_lds_b128 %0, %1, off"
                         :: "v"(l), "v"((unsigned long long)(uintptr_t)gp)
                         : "memory");
        }
    };

    issueA(0, 0);
    issueB(0, 0);
    asm volatile("s_wait_asynccnt 0x0" ::: "memory");
    __syncthreads();

    int buf = 0;
    for (int k0 = 0; k0 < K; k0 += BK) {
        int kn = k0 + BK;
        if (kn < K) {                      // prefetch next tile into other buf
            issueA(kn, buf ^ 1);
            issueB(kn, buf ^ 1);
        }

        // fragment loads: lane<16 -> K {kh..kh+7, kh+16..kh+23}, kh=8*(lane>=16)
        v16h af[2];
        #pragma unroll
        for (int mi = 0; mi < 2; mi++) {
            int arow = 32 * wr + 16 * mi + lrow;
            v8h lo = *(const v8h*)&lA[buf][arow][khalf];
            v8h hi = *(const v8h*)&lA[buf][arow][khalf + 16];
            af[mi] = __builtin_shufflevector(lo, hi,
                0,1,2,3,4,5,6,7,8,9,10,11,12,13,14,15);
        }
        v16h bfr[4];
        #pragma unroll
        for (int ni = 0; ni < 4; ni++) {
            int bcol = 64 * wc + 16 * ni + lrow;
            v8h lo = *(const v8h*)&lB[buf][bcol][khalf];
            v8h hi = *(const v8h*)&lB[buf][bcol][khalf + 16];
            bfr[ni] = __builtin_shufflevector(lo, hi,
                0,1,2,3,4,5,6,7,8,9,10,11,12,13,14,15);
        }
        #pragma unroll
        for (int mi = 0; mi < 2; mi++)
            #pragma unroll
            for (int ni = 0; ni < 4; ni++)
                acc[mi][ni] = __builtin_amdgcn_wmma_f32_16x16x32_f16(
                    false, af[mi], false, bfr[ni], (short)0,
                    acc[mi][ni], false, false);

        asm volatile("s_wait_asynccnt 0x0" ::: "memory");
        __syncthreads();
        buf ^= 1;
    }

    // --- epilogue: C layout: VGPR v -> row v + 8*(lane>=16), col lane&15 ---
    const int rbase = (lane >> 4) * 8;
    const int cN    = lane & 15;
    #pragma unroll
    for (int mi = 0; mi < 2; mi++)
        #pragma unroll
        for (int ni = 0; ni < 4; ni++) {
            int gc = n0 + 64 * wc + 16 * ni + cN;
            float bv = bias[gc];
            #pragma unroll
            for (int v = 0; v < 8; v++) {
                int gr = m0 + 32 * wr + 16 * mi + rbase + v;
                if (gr < M) {
                    float val = acc[mi][ni][v] + bv;
                    if (ACT) val = gelu_f(val);
                    Cout[(long)gr * N_ + gc] = (_Float16)val;
                }
            }
        }
}

// ---------------------------------------------------------------------------
// f32 -> f16 bulk convert (n divisible by 1024)
// ---------------------------------------------------------------------------
__global__ void cvt_f32_to_f16(const float* __restrict__ in,
                               _Float16* __restrict__ out)
{
    long i = ((long)blockIdx.x * blockDim.x + threadIdx.x) * 4;
    v4f v = *(const v4f*)(in + i);
    v4h h;
    #pragma unroll
    for (int j = 0; j < 4; j++) h[j] = (_Float16)v[j];
    *(v4h*)(out + i) = h;
}

// Wt[n*K + k] = (f16) W[k*N + n]   (transpose + convert; N*K divisible by 256)
__global__ void transpose_to_f16(const float* __restrict__ W,
                                 _Float16* __restrict__ Wt, int K, int N)
{
    int idx = blockIdx.x * 256 + threadIdx.x;
    int n = idx / K, k = idx % K;
    Wt[idx] = (_Float16)W[(long)k * N + n];
}

// ---------------------------------------------------------------------------
// LayerNorm over D=512 (f16 in), writing f16 into nodes[g, 1+s, :]
// ---------------------------------------------------------------------------
__global__ __launch_bounds__(256) void ln_concat(
    const _Float16* __restrict__ H2, const float* __restrict__ lg,
    const float* __restrict__ lb, _Float16* __restrict__ nodes)
{
    __shared__ float sb[256];
    __shared__ float stat[2];
    int t = threadIdx.x;
    long row = blockIdx.x;
    long off = row * DDIM;
    float x0 = (float)H2[off + t], x1 = (float)H2[off + t + 256];
    sb[t] = x0 + x1;
    __syncthreads();
    for (int st = 128; st >= 1; st >>= 1) {
        if (t < st) sb[t] += sb[t + st];
        __syncthreads();
    }
    if (t == 0) stat[0] = sb[0] * (1.0f / (float)DDIM);
    __syncthreads();
    float mean = stat[0];
    float d0 = x0 - mean, d1 = x1 - mean;
    sb[t] = d0 * d0 + d1 * d1;
    __syncthreads();
    for (int st = 128; st >= 1; st >>= 1) {
        if (t < st) sb[t] += sb[t + st];
        __syncthreads();
    }
    if (t == 0) stat[1] = rsqrtf(sb[0] * (1.0f / (float)DDIM) + 1e-5f);
    __syncthreads();
    float rstd = stat[1];
    int g  = (int)(row >> 11);      // row / SEQ
    int si = (int)(row & 2047);     // row % SEQ
    long nrow = ((long)g * NNODE + 1 + si) * DDIM;
    nodes[nrow + t]       = (_Float16)(d0 * rstd * lg[t]       + lb[t]);
    nodes[nrow + t + 256] = (_Float16)(d1 * rstd * lg[t + 256] + lb[t + 256]);
}

__global__ void copy_pooled(const float* __restrict__ pooled,
                            _Float16* __restrict__ nodes)
{
    int g = blockIdx.x, t = threadIdx.x;
    long dst = (long)g * NNODE * DDIM;
    nodes[dst + t]       = (_Float16)pooled[g * DDIM + t];
    nodes[dst + t + 256] = (_Float16)pooled[g * DDIM + t + 256];
}

// ---------------------------------------------------------------------------
// GATv2 on star graph: non-root node j gets edges {0->j, j->j}: 2-way softmax
// ---------------------------------------------------------------------------
__global__ __launch_bounds__(256) void gat_nonroot(
    const _Float16* __restrict__ xl, const _Float16* __restrict__ xr,
    const float* __restrict__ att, const float* __restrict__ bias,
    _Float16* __restrict__ out)
{
    __shared__ float r0[256], r1[256];
    __shared__ float w0s[NHEAD], w1s[NHEAD];
    int t = threadIdx.x;
    int e = blockIdx.x;                       // 0 .. B*(N-1)-1
    int g = e / (NNODE - 1);
    int j = e % (NNODE - 1) + 1;
    long root = (long)(g * NNODE) * HCDIM;
    long idx  = (long)(g * NNODE + j) * HCDIM;
    float xl0 = (float)xl[root + t];
    float xlj = (float)xl[idx + t];
    float xrj = (float)xr[idx + t];
    float av  = att[t];
    r0[t] = lrelu_f(xl0 + xrj) * av;
    r1[t] = lrelu_f(xlj + xrj) * av;
    __syncthreads();
    for (int s = 32; s >= 1; s >>= 1) {       // reduce within 64-ch head group
        if ((t & 63) < s) { r0[t] += r0[t + s]; r1[t] += r1[t + s]; }
        __syncthreads();
    }
    int h = t >> 6;
    if ((t & 63) == 0) {
        float l0 = r0[t], l1 = r1[t];
        float m  = fmaxf(l0, l1);
        float a0 = expf(l0 - m), a1 = expf(l1 - m);
        float dn = a0 + a1 + 1e-16f;
        w0s[h] = a0 / dn;
        w1s[h] = a1 / dn;
    }
    __syncthreads();
    float v = w0s[h] * xl0 + w1s[h] * xlj + bias[t];
    out[idx + t] = (_Float16)gelu_f(v);
}

// Root node: online softmax over all N sources (j->0 for j>=1, plus self 0->0)
__global__ __launch_bounds__(256) void gat_root(
    const _Float16* __restrict__ xl, const _Float16* __restrict__ xr,
    const float* __restrict__ att, const float* __restrict__ bias,
    _Float16* __restrict__ out)
{
    __shared__ float red[256];
    int t = threadIdx.x, g = blockIdx.x;
    int h = t >> 6;
    long base = (long)g * NNODE * HCDIM;
    float xr0 = (float)xr[base + t];
    float av  = att[t];
    float m = -1e30f, d = 0.0f, acc = 0.0f;
    for (int s = 0; s < NNODE; s++) {
        float xls = (float)xl[base + (long)s * HCDIM + t];
        red[t] = lrelu_f(xls + xr0) * av;
        __syncthreads();
        for (int st = 32; st >= 1; st >>= 1) {
            if ((t & 63) < st) red[t] += red[t + st];
            __syncthreads();
        }
        float l = red[h * 64];                 // per-head logit, broadcast
        float mn = fmaxf(m, l);
        float sc = expf(m - mn), p = expf(l - mn);
        d   = d * sc + p;
        acc = acc * sc + p * xls;
        m = mn;
        __syncthreads();                        // before red is overwritten
    }
    out[base + t] = (_Float16)gelu_f(acc / (d + 1e-16f) + bias[t]);
}

// ---------------------------------------------------------------------------
// Mean over nodes per graph (f16 in, f32 out)
// ---------------------------------------------------------------------------
__global__ void graph_mean_k(const _Float16* __restrict__ x,
                             float* __restrict__ outp)
{
    int t = threadIdx.x, g = blockIdx.x;
    long base = (long)g * NNODE * HCDIM;
    float s = 0.0f;
    for (int i = 0; i < NNODE; i++) s += (float)x[base + (long)i * HCDIM + t];
    outp[g * HCDIM + t] = s * (1.0f / (float)NNODE);
}

// ---------------------------------------------------------------------------
// Small FC: Y[b,o] = act(dot(X[b,:K], W[:,o]) + bias[o] (+ residual[b,o]))
// ---------------------------------------------------------------------------
template<typename TX>
__global__ void fc_kernel(const TX* __restrict__ X, const float* __restrict__ W,
                          const float* __restrict__ bias, float* __restrict__ Y,
                          int Kdim, int Ndim, long x_row_stride, int act,
                          const float* __restrict__ residual)
{
    int lin = blockIdx.x * blockDim.x + threadIdx.x;
    int b = lin / Ndim, o = lin % Ndim;
    const TX* xrow = X + (long)b * x_row_stride;
    float s = bias[o];
    for (int k = 0; k < Kdim; k++) s += (float)xrow[k] * W[(long)k * Ndim + o];
    if (residual) s += residual[(long)b * Ndim + o];
    if (act) s = gelu_f(s);
    Y[(long)b * Ndim + o] = s;
}

// ---------------------------------------------------------------------------
// Launcher
// ---------------------------------------------------------------------------
extern "C" void kernel_launch(void* const* d_in, const int* in_sizes, int n_in,
                              void* d_out, int out_size, void* d_ws, size_t ws_size,
                              hipStream_t stream)
{
    const float* hs     = (const float*)d_in[0];
    const float* pooled = (const float*)d_in[1];
    const float* pW1    = (const float*)d_in[2];
    const float* pb1    = (const float*)d_in[3];
    const float* pW2    = (const float*)d_in[4];
    const float* pb2    = (const float*)d_in[5];
    const float* ln_g   = (const float*)d_in[6];
    const float* ln_b   = (const float*)d_in[7];
    const float* wl1    = (const float*)d_in[8];
    const float* bl1    = (const float*)d_in[9];
    const float* wr1    = (const float*)d_in[10];
    const float* br1    = (const float*)d_in[11];
    const float* att1   = (const float*)d_in[12];
    const float* bias1  = (const float*)d_in[13];
    const float* wl2    = (const float*)d_in[14];
    const float* bl2    = (const float*)d_in[15];
    const float* wr2    = (const float*)d_in[16];
    const float* br2    = (const float*)d_in[17];
    const float* att2   = (const float*)d_in[18];
    const float* bias2  = (const float*)d_in[19];
    const float* mW1    = (const float*)d_in[20];
    const float* mb1    = (const float*)d_in[21];
    const float* mW2    = (const float*)d_in[22];
    const float* mb2    = (const float*)d_in[23];
    const float* fW     = (const float*)d_in[24];
    const float* fb     = (const float*)d_in[25];

    // ---- f16 workspace layout (all sizes multiple of 8 halfs) -------------
    _Float16* wsh = (_Float16*)d_ws;
    long off = 0;
    _Float16* hs_h    = wsh + off; off += (long)NROWS * FT_IN;   // 100,663,296
    _Float16* H1h     = wsh + off;                                // reused:
    _Float16* nodes_h = H1h;       off += (long)NTOT * DDIM;      // 67,141,632
    _Float16* H2h     = wsh + off; off += (long)NROWS * DDIM;     // 67,108,864
    _Float16* xl_h    = wsh + off; off += (long)NTOT * HCDIM;
    _Float16* xr_h    = wsh + off; off += (long)NTOT * HCDIM;
    _Float16* xc_h    = wsh + off; off += (long)NTOT * HCDIM;
    _Float16* pW1t    = wsh + off; off += (long)FT_IN * DDIM;
    _Float16* pW2t    = wsh + off; off += (long)DDIM * DDIM;
    _Float16* wl1t    = wsh + off; off += (long)DDIM * HCDIM;
    _Float16* wr1t    = wsh + off; off += (long)DDIM * HCDIM;
    _Float16* wl2t    = wsh + off; off += (long)HCDIM * HCDIM;
    _Float16* wr2t    = wsh + off; off += (long)HCDIM * HCDIM;
    float* r1 = (float*)(wsh + off);          // 64*1024 f32
    float* r2 = r1 + 64 * 1024;               // 64*256  f32

    float* out_main = (float*)d_out;                  // (B, D)
    float* out_mean = (float*)d_out + BATCH * DDIM;   // (B, HC)

    dim3 blk(256);

    // 0) convert inputs to f16 / pre-transpose weights
    cvt_f32_to_f16<<<((long)NROWS * FT_IN) / 1024, 256, 0, stream>>>(hs, hs_h);
    transpose_to_f16<<<(FT_IN * DDIM) / 256, 256, 0, stream>>>(pW1, pW1t, FT_IN, DDIM);
    transpose_to_f16<<<(DDIM * DDIM) / 256, 256, 0, stream>>>(pW2, pW2t, DDIM, DDIM);
    transpose_to_f16<<<(DDIM * HCDIM) / 256, 256, 0, stream>>>(wl1, wl1t, DDIM, HCDIM);
    transpose_to_f16<<<(DDIM * HCDIM) / 256, 256, 0, stream>>>(wr1, wr1t, DDIM, HCDIM);
    transpose_to_f16<<<(HCDIM * HCDIM) / 256, 256, 0, stream>>>(wl2, wl2t, HCDIM, HCDIM);
    transpose_to_f16<<<(HCDIM * HCDIM) / 256, 256, 0, stream>>>(wr2, wr2t, HCDIM, HCDIM);

    // 1) h = gelu(hs @ pW1 + pb1)   [131072 x 768 x 512]
    gemm_wmma<1><<<dim3(DDIM / BN, NROWS / BM), blk, 0, stream>>>(
        hs_h, pW1t, pb1, H1h, NROWS, FT_IN, DDIM);
    // 2) h = gelu(h @ pW2 + pb2)    [131072 x 512 x 512]
    gemm_wmma<1><<<dim3(DDIM / BN, NROWS / BM), blk, 0, stream>>>(
        H1h, pW2t, pb2, H2h, NROWS, DDIM, DDIM);
    // 3) layernorm + concat into nodes (H1 region is dead now)
    ln_concat<<<NROWS, 256, 0, stream>>>(H2h, ln_g, ln_b, nodes_h);
    copy_pooled<<<BATCH, 256, 0, stream>>>(pooled, nodes_h);

    // 4) GAT layer 1 projections  [131136 x 512 x 256]
    dim3 gp(HCDIM / BN, (NTOT + BM - 1) / BM);
    gemm_wmma<0><<<gp, blk, 0, stream>>>(nodes_h, wl1t, bl1, xl_h, NTOT, DDIM, HCDIM);
    gemm_wmma<0><<<gp, blk, 0, stream>>>(nodes_h, wr1t, br1, xr_h, NTOT, DDIM, HCDIM);
    // 5) GAT layer 1 aggregation -> xc
    gat_nonroot<<<BATCH * (NNODE - 1), 256, 0, stream>>>(xl_h, xr_h, att1, bias1, xc_h);
    gat_root<<<BATCH, 256, 0, stream>>>(xl_h, xr_h, att1, bias1, xc_h);

    // 6) GAT layer 2 projections  [131136 x 256 x 256]
    gemm_wmma<0><<<gp, blk, 0, stream>>>(xc_h, wl2t, bl2, xl_h, NTOT, HCDIM, HCDIM);
    gemm_wmma<0><<<gp, blk, 0, stream>>>(xc_h, wr2t, br2, xr_h, NTOT, HCDIM, HCDIM);
    // 7) GAT layer 2 aggregation -> xc (inputs xl/xr only; x1 is dead)
    gat_nonroot<<<BATCH * (NNODE - 1), 256, 0, stream>>>(xl_h, xr_h, att2, bias2, xc_h);
    gat_root<<<BATCH, 256, 0, stream>>>(xl_h, xr_h, att2, bias2, xc_h);

    // 8) graph mean -> second output
    graph_mean_k<<<BATCH, 256, 0, stream>>>(xc_h, out_mean);

    // 9) head MLP on root rows (row stride N*HC)
    fc_kernel<_Float16><<<(BATCH * 1024) / 256, 256, 0, stream>>>(
        xc_h, mW1, mb1, r1, HCDIM, 4 * HCDIM, (long)NNODE * HCDIM, 1, nullptr);
    fc_kernel<float><<<(BATCH * HCDIM) / 256, 256, 0, stream>>>(
        r1, mW2, mb2, r2, 4 * HCDIM, HCDIM, 4 * HCDIM, 1, nullptr);
    fc_kernel<float><<<(BATCH * DDIM) / 256, 256, 0, stream>>>(
        r2, fW, fb, out_main, HCDIM, DDIM, HCDIM, 0, pooled);
}
